// MADFormer_10746008175040
// MI455X (gfx1250) — matrix operand order repaired and checked by reading, not compiled, not run on hardware
//
#include <hip/hip_runtime.h>
#include <hip/hip_bf16.h>

// ---------------------------------------------------------------------------
// MADFormer layer for MI455X (gfx1250, wave32, WMMA bf16 16x16x32 + TDM).
// B=2, N=1280 (256 text + 512 + 512 image), C=1024, H=16, Dh=64, I=4096.
// ---------------------------------------------------------------------------

typedef __attribute__((ext_vector_type(16))) __bf16 v16bf;
typedef __attribute__((ext_vector_type(8)))  __bf16 v8bf;
typedef __attribute__((ext_vector_type(8)))  float  v8f;
typedef unsigned int v4u __attribute__((ext_vector_type(4)));
typedef unsigned int v8u __attribute__((ext_vector_type(8)));

#define BB     2
#define NN     1280
#define CC     1024
#define HH     16
#define DH     64
#define II     4096
#define TLEN   256
#define BLKSZ  128
#define LEN1   4
#define MM     (BB * NN)      // 2560 rows
#define C3     (3 * CC)       // 3072

static __device__ __forceinline__ int seg_of(int nseq) {
  return nseq < TLEN ? 0 : (nseq < TLEN + LEN1 * BLKSZ ? 1 : 2);
}

static __device__ __forceinline__ __bf16 f2bf(float f) {
  __hip_bfloat16 h = __float2bfloat16(f);
  __bf16 r;
  __builtin_memcpy(&r, &h, sizeof(r));
  return r;
}

static __device__ __forceinline__ v8f wmma_bf16(v16bf a, v16bf b, v8f c) {
  return __builtin_amdgcn_wmma_f32_16x16x32_bf16(false, a, false, b,
                                                 (short)0, c, false, false);
}

// A-fragment (16x32 bf16): lane holds row (lane&15); elements 0..15 are
// K = base + half*16 .. +15.
// `p` must already point at rowBase + half*16.
static __device__ __forceinline__ v16bf make_afrag(const __bf16* p) {
  v8bf lo = *(const v8bf*)(p);
  v8bf hi = *(const v8bf*)(p + 8);
  v16bf a;
#pragma unroll
  for (int i = 0; i < 8; ++i) { a[i] = lo[i]; a[i + 8] = hi[i]; }
  return a;
}

// ---------------------------------------------------------------------------
// Tensor Data Mover: async-DMA a 32x64 bf16 tile (row stride 64 elems in
// global) into LDS with +16-elem row padding (pad 8 DWORDs per 32 DWORDs),
// giving an 80-element LDS row stride. Tracked by TENSORcnt.
// D# per CDNA5 ISA ch.8: group0 = {ctrl, lds_addr, gaddr_lo, gaddr_hi|type},
// group1 = {flags, dims/tile/strides} (2D tile -> groups 2/3 omitted).
// ---------------------------------------------------------------------------
static __device__ __forceinline__ void tdm_load_vtile(const __bf16* gsrc,
                                                      __bf16* lds_dst) {
  unsigned long long ga = (unsigned long long)(uintptr_t)gsrc;
  unsigned int       la = (unsigned int)(uintptr_t)lds_dst;
  v4u g0;
  g0[0] = 1u;                                   // count=1, user load D#
  g0[1] = la;                                   // lds_addr (bytes)
  g0[2] = (unsigned int)ga;                     // global_addr[31:0]
  g0[3] = ((unsigned int)(ga >> 32) & 0x01FFFFFFu) | 0x80000000u; // type=2
  v8u g1;
  g1[0] = (1u << 16)        // data_size = 2 bytes (bf16)
        | (1u << 20)        // pad_enable
        | (4u << 22)        // pad_interval: 32 DWORDs (one 64-elem row)
        | (7u << 25);       // pad_amount: 8 DWORDs (16 bf16 elems)
  g1[1] = (64u & 0xFFFFu) << 16;        // tensor_dim0 = 64  (bits [79:48])
  g1[2] = (1280u & 0xFFFFu) << 16;      // tensor_dim1 = N   (bits [111:80])
  g1[3] = (64u << 16);                  // tile_dim0 = 64    (bits [127:112])
  g1[4] = 32u;                          // tile_dim1 = 32, tile_dim2 = 0
  g1[5] = 64u;                          // tensor_dim0_stride = 64 elems
  g1[6] = 0u;
  g1[7] = 0u;
  asm volatile("tensor_load_to_lds %0, %1" :: "s"(g0), "s"(g1) : "memory");
}

// ---------------------------------------------------------------------------
// Weight transpose + fp32 -> bf16. src: [3][R][Cc] f32 -> dst: [3][Cc][R] bf16
// ---------------------------------------------------------------------------
__global__ __launch_bounds__(256) void k_transpose_bf16(
    const float* __restrict__ src, __bf16* __restrict__ dst, int R, int Cc) {
  size_t per = (size_t)R * Cc;
  size_t total = 3 * per;
  size_t i = (size_t)blockIdx.x * 256 + threadIdx.x;
  if (i >= total) return;
  int s   = (int)(i / per);
  size_t rem = i - (size_t)s * per;
  int r = (int)(rem / Cc);
  int c = (int)(rem - (size_t)r * Cc);
  dst[(size_t)s * per + (size_t)c * R + r] = f2bf(src[i]);
}

// ---------------------------------------------------------------------------
// Row RMSNorm (C=1024), per-segment weight, fp32 in -> bf16 out.
// ---------------------------------------------------------------------------
__global__ __launch_bounds__(256) void k_rmsnorm(
    const float* __restrict__ src, const float* __restrict__ w,
    __bf16* __restrict__ dst) {
  __shared__ float red[256];
  int row  = blockIdx.x;
  int seg  = seg_of(row % NN);
  int t    = threadIdx.x;
  const float* p = src + (size_t)row * CC;
  float4 vv = *(const float4*)(p + t * 4);
  float s = vv.x * vv.x + vv.y * vv.y + vv.z * vv.z + vv.w * vv.w;
  red[t] = s;
  __syncthreads();
#pragma unroll
  for (int st = 128; st > 0; st >>= 1) {
    if (t < st) red[t] += red[t + st];
    __syncthreads();
  }
  float scale = rsqrtf(red[0] / (float)CC + 1e-6f);
  const float* ws = w + seg * CC;
  __bf16* d = dst + (size_t)row * CC + t * 4;
  d[0] = f2bf(vv.x * ws[t * 4 + 0] * scale);
  d[1] = f2bf(vv.y * ws[t * 4 + 1] * scale);
  d[2] = f2bf(vv.z * ws[t * 4 + 2] * scale);
  d[3] = f2bf(vv.w * ws[t * 4 + 3] * scale);
}

// ---------------------------------------------------------------------------
// QKV GEMM: out[M][3C] f32 = xn[M][C] @ Wt[seg][3C][C]^T + bias[seg][3C]
// One wave per 16x64 output tile.
// ---------------------------------------------------------------------------
__global__ __launch_bounds__(32) void k_gemm_qkv(
    const __bf16* __restrict__ xn, const __bf16* __restrict__ wT,
    const float* __restrict__ bias, float* __restrict__ out) {
  const int lane = threadIdx.x & 31;
  const int half = lane >> 4, lx = lane & 15;
  const int n0 = blockIdx.x * 64;
  const int m0 = blockIdx.y * 16;
  const int seg = seg_of(m0 % NN);
  const __bf16* W = wT + (size_t)seg * C3 * CC;
  const float*  bs = bias + (size_t)seg * C3;

  v8f acc[4] = {};
  const __bf16* Arow = xn + (size_t)(m0 + lx) * CC + half * 16;
  for (int k0 = 0; k0 < CC; k0 += 32) {
    v16bf a = make_afrag(Arow + k0);
    if (k0 + 32 < CC)
      __builtin_prefetch(W + (size_t)(n0 + lx) * CC + k0 + 32, 0, 1);
#pragma unroll
    for (int t = 0; t < 4; ++t) {
      const __bf16* Br = W + (size_t)(n0 + t * 16 + lx) * CC + k0 + half * 16;
      v16bf b = *(const v16bf*)Br;
      acc[t] = wmma_bf16(a, b, acc[t]);
    }
  }
#pragma unroll
  for (int t = 0; t < 4; ++t) {
    int col = n0 + t * 16 + lx;
    float bv = bs[col];
#pragma unroll
    for (int r = 0; r < 8; ++r)
      out[(size_t)(m0 + half * 8 + r) * C3 + col] = acc[t][r] + bv;
  }
}

// ---------------------------------------------------------------------------
// Per-head q/k RMSNorm + RoPE + qkv split into [B*H][N][64] bf16.
// One wave per (b,h,n); lane t owns the rotation pair (t, t+32).
// ---------------------------------------------------------------------------
__global__ __launch_bounds__(32) void k_qk_rope(
    const float* __restrict__ qkv, const int* __restrict__ ipos,
    const int* __restrict__ tpos, const float* __restrict__ qn_w,
    const float* __restrict__ kn_w, __bf16* __restrict__ qo,
    __bf16* __restrict__ ko, __bf16* __restrict__ vo) {
  int gid = blockIdx.x;                 // (b*H + h)*N + n
  int n = gid % NN;
  int bh = gid / NN;
  int seg = seg_of(n);
  int t = threadIdx.x;
  int b = bh / HH, h = bh % HH;

  size_t base = (size_t)(b * NN + n) * C3 + h * DH;
  float q1 = qkv[base + t],        q2 = qkv[base + t + 32];
  float k1 = qkv[base + CC + t],   k2 = qkv[base + CC + t + 32];
  float v1 = qkv[base + 2*CC + t], v2 = qkv[base + 2*CC + t + 32];

  float sq = q1 * q1 + q2 * q2;
  float sk = k1 * k1 + k2 * k2;
#pragma unroll
  for (int off = 16; off > 0; off >>= 1) {
    sq += __shfl_xor(sq, off, 32);
    sk += __shfl_xor(sk, off, 32);
  }
  float rq = rsqrtf(sq / (float)DH + 1e-6f);
  float rk = rsqrtf(sk / (float)DH + 1e-6f);
  q1 *= rq * qn_w[seg * DH + t];
  q2 *= rq * qn_w[seg * DH + t + 32];
  k1 *= rk * kn_w[seg * DH + t];
  k2 *= rk * kn_w[seg * DH + t + 32];

  int pos = (n < TLEN) ? tpos[n] : ipos[n - TLEN];
  // inv_freq[t] = 10000^(-2t/64)
  float ang = (float)pos * __expf(-(float)(2 * t) * (9.2103403719761836f / 64.0f));
  float c = __cosf(ang), s = __sinf(ang);

  size_t ob = ((size_t)bh * NN + n) * DH;
  qo[ob + t]      = f2bf(q1 * c - q2 * s);
  qo[ob + t + 32] = f2bf(q2 * c + q1 * s);
  ko[ob + t]      = f2bf(k1 * c - k2 * s);
  ko[ob + t + 32] = f2bf(k2 * c + k1 * s);
  vo[ob + t]      = f2bf(v1);
  vo[ob + t + 32] = f2bf(v2);
}

// ---------------------------------------------------------------------------
// Skip-causal mask from the reference.
// ---------------------------------------------------------------------------
static __device__ __forceinline__ bool attn_mask(int qi, int ki) {
  if (ki < TLEN) return qi >= ki;
  if (qi < TLEN) return false;
  int qb = (qi - TLEN) / BLKSZ;
  int kb = (ki - TLEN) / BLKSZ;
  if (kb < LEN1) {
    if (qb < LEN1) return qb >= kb;
    return (qb - LEN1) > kb;
  }
  return qb == kb;
}

// ---------------------------------------------------------------------------
// Flash attention per (b,h, 16-row q tile); 32-key steps, WMMA throughout.
// V tiles are DMA'd Global->LDS by the Tensor Data Mover, double-buffered so
// tile i+1 streams in while tile i is consumed. Output written directly in
// [B][N][C] bf16 layout for the proj GEMM.
// ---------------------------------------------------------------------------
__global__ __launch_bounds__(32) void k_attention(
    const __bf16* __restrict__ q, const __bf16* __restrict__ k,
    const __bf16* __restrict__ v, __bf16* __restrict__ o) {
  __shared__ __align__(32) __bf16 lds_p[16 * 32];
  __shared__ __align__(32) __bf16 lds_v[2][32 * 80]; // 32 keys x 64, pad to 80

  const int gid  = blockIdx.x;        // (b*H + h)*80 + qt
  const int qt   = gid % (NN / 16);
  const int bh   = gid / (NN / 16);
  const int lane = threadIdx.x & 31;
  const int half = lane >> 4, lx = lane & 15;
  const size_t hb = (size_t)bh * NN * DH;

  const __bf16* qrow = q + hb + (size_t)(qt * 16 + lx) * DH + half * 16;
  v16bf aq0 = make_afrag(qrow);
  v16bf aq1 = make_afrag(qrow + 32);

  v8f oacc[4] = {};
  float mrow[8], lrow[8];
#pragma unroll
  for (int r = 0; r < 8; ++r) { mrow[r] = -1e30f; lrow[r] = 0.0f; }
  const float scale = 0.125f;   // 1/sqrt(64)

  // Kick off the TDM for the first V tile.
  tdm_load_vtile(v + hb, &lds_v[0][0]);

  for (int kb0 = 0; kb0 < NN; kb0 += 32) {
    const int buf = (kb0 >> 5) & 1;
    // Prefetch next V tile via TDM while we work on this one.
    if (kb0 + 32 < NN)
      tdm_load_vtile(v + hb + (size_t)(kb0 + 32) * DH, &lds_v[buf ^ 1][0]);

    // --- S = Q @ K^T for 32 keys ---
    const __bf16* kr0 = k + hb + (size_t)(kb0 + lx) * DH;
    const __bf16* kr1 = k + hb + (size_t)(kb0 + 16 + lx) * DH;
    v8f s0 = {}, s1 = {};
    s0 = wmma_bf16(aq0, *(const v16bf*)(kr0 + half * 16), s0);
    s0 = wmma_bf16(aq1, *(const v16bf*)(kr0 + 32 + half * 16), s0);
    s1 = wmma_bf16(aq0, *(const v16bf*)(kr1 + half * 16), s1);
    s1 = wmma_bf16(aq1, *(const v16bf*)(kr1 + 32 + half * 16), s1);

    // --- mask + online softmax (row stats per lane for its 8 rows) ---
    float p0[8], p1[8];
#pragma unroll
    for (int r = 0; r < 8; ++r) {
      int qi = qt * 16 + half * 8 + r;
      float x0 = s0[r] * scale;
      float x1 = s1[r] * scale;
      if (!attn_mask(qi, kb0 + lx))      x0 = -1e30f;
      if (!attn_mask(qi, kb0 + 16 + lx)) x1 = -1e30f;
      float tm = fmaxf(x0, x1);
#pragma unroll
      for (int off = 1; off < 16; off <<= 1)
        tm = fmaxf(tm, __shfl_xor(tm, off, 32));
      float nm = fmaxf(mrow[r], tm);
      float al = __expf(mrow[r] - nm);
      float e0 = __expf(x0 - nm);
      float e1 = __expf(x1 - nm);
      float rs = e0 + e1;
#pragma unroll
      for (int off = 1; off < 16; off <<= 1)
        rs += __shfl_xor(rs, off, 32);
      lrow[r] = lrow[r] * al + rs;
      mrow[r] = nm;
      p0[r] = e0;
      p1[r] = e1;
#pragma unroll
      for (int t = 0; t < 4; ++t) oacc[t][r] *= al;
    }

    // --- reshape P: C-layout regs -> LDS -> A-fragment ---
#pragma unroll
    for (int r = 0; r < 8; ++r) {
      lds_p[(half * 8 + r) * 32 + lx]      = f2bf(p0[r]);
      lds_p[(half * 8 + r) * 32 + 16 + lx] = f2bf(p1[r]);
    }
    __syncthreads();
    v16bf pa = make_afrag(&lds_p[lx * 32 + half * 16]);

    // Current V tile must have landed (next one may still be in flight).
    if (kb0 + 32 < NN) __builtin_amdgcn_s_wait_tensorcnt(1);
    else               __builtin_amdgcn_s_wait_tensorcnt(0);

    // --- O += P @ V ---
#pragma unroll
    for (int t = 0; t < 4; ++t) {
      v16bf bv;
#pragma unroll
      for (int j = 0; j < 16; ++j)
        bv[j] = lds_v[buf][(half * 16 + j) * 80 + t * 16 + lx];
      oacc[t] = wmma_bf16(pa, bv, oacc[t]);
    }
    __syncthreads();
  }

  // --- normalize + write [B][N][C] bf16 ---
  const int b = bh / HH, h = bh % HH;
#pragma unroll
  for (int r = 0; r < 8; ++r) {
    int qi = qt * 16 + half * 8 + r;
    float inv = 1.0f / lrow[r];
#pragma unroll
    for (int t = 0; t < 4; ++t)
      o[(size_t)(b * NN + qi) * CC + h * DH + t * 16 + lx] =
          f2bf(oacc[t][r] * inv);
  }
}

// ---------------------------------------------------------------------------
// Proj GEMM + residual: h[M][C] = x + o @ Wt[seg]^T + bias[seg]
// ---------------------------------------------------------------------------
__global__ __launch_bounds__(32) void k_gemm_proj(
    const __bf16* __restrict__ ob, const __bf16* __restrict__ wT,
    const float* __restrict__ bias, const float* __restrict__ x,
    float* __restrict__ h) {
  const int lane = threadIdx.x & 31;
  const int half = lane >> 4, lx = lane & 15;
  const int n0 = blockIdx.x * 64;
  const int m0 = blockIdx.y * 16;
  const int seg = seg_of(m0 % NN);
  const __bf16* W = wT + (size_t)seg * CC * CC;
  const float*  bs = bias + (size_t)seg * CC;

  v8f acc[4] = {};
  const __bf16* Arow = ob + (size_t)(m0 + lx) * CC + half * 16;
  for (int k0 = 0; k0 < CC; k0 += 32) {
    v16bf a = make_afrag(Arow + k0);
    if (k0 + 32 < CC)
      __builtin_prefetch(W + (size_t)(n0 + lx) * CC + k0 + 32, 0, 1);
#pragma unroll
    for (int t = 0; t < 4; ++t) {
      const __bf16* Br = W + (size_t)(n0 + t * 16 + lx) * CC + k0 + half * 16;
      acc[t] = wmma_bf16(a, *(const v16bf*)Br, acc[t]);
    }
  }
#pragma unroll
  for (int t = 0; t < 4; ++t) {
    int col = n0 + t * 16 + lx;
    float bv = bs[col];
#pragma unroll
    for (int r = 0; r < 8; ++r) {
      size_t idx = (size_t)(m0 + half * 8 + r) * CC + col;
      h[idx] = x[idx] + acc[t][r] + bv;
    }
  }
}

// ---------------------------------------------------------------------------
// Gate/Up GEMM fused: act[M][I] bf16 = silu(yn@Gt^T) * (yn@Ut^T)
// ---------------------------------------------------------------------------
__global__ __launch_bounds__(32) void k_gemm_gateup(
    const __bf16* __restrict__ yn, const __bf16* __restrict__ gT,
    const __bf16* __restrict__ uT, __bf16* __restrict__ act) {
  const int lane = threadIdx.x & 31;
  const int half = lane >> 4, lx = lane & 15;
  const int n0 = blockIdx.x * 64;
  const int m0 = blockIdx.y * 16;
  const int seg = seg_of(m0 % NN);
  const __bf16* G = gT + (size_t)seg * II * CC;
  const __bf16* U = uT + (size_t)seg * II * CC;

  v8f ag[4] = {}, au[4] = {};
  const __bf16* Arow = yn + (size_t)(m0 + lx) * CC + half * 16;
  for (int k0 = 0; k0 < CC; k0 += 32) {
    v16bf a = make_afrag(Arow + k0);
    if (k0 + 32 < CC) {
      __builtin_prefetch(G + (size_t)(n0 + lx) * CC + k0 + 32, 0, 1);
      __builtin_prefetch(U + (size_t)(n0 + lx) * CC + k0 + 32, 0, 1);
    }
#pragma unroll
    for (int t = 0; t < 4; ++t) {
      size_t roff = (size_t)(n0 + t * 16 + lx) * CC + k0 + half * 16;
      ag[t] = wmma_bf16(a, *(const v16bf*)(G + roff), ag[t]);
      au[t] = wmma_bf16(a, *(const v16bf*)(U + roff), au[t]);
    }
  }
#pragma unroll
  for (int t = 0; t < 4; ++t) {
    int col = n0 + t * 16 + lx;
#pragma unroll
    for (int r = 0; r < 8; ++r) {
      float g = ag[t][r];
      float s = g / (1.0f + __expf(-g));   // silu
      act[(size_t)(m0 + half * 8 + r) * II + col] = f2bf(s * au[t][r]);
    }
  }
}

// ---------------------------------------------------------------------------
// Down GEMM + residual: out[M][C] f32 = h + act @ Dt[seg]^T
// ---------------------------------------------------------------------------
__global__ __launch_bounds__(32) void k_gemm_down(
    const __bf16* __restrict__ act, const __bf16* __restrict__ dT,
    const float* __restrict__ h, float* __restrict__ out) {
  const int lane = threadIdx.x & 31;
  const int half = lane >> 4, lx = lane & 15;
  const int n0 = blockIdx.x * 64;
  const int m0 = blockIdx.y * 16;
  const int seg = seg_of(m0 % NN);
  const __bf16* W = dT + (size_t)seg * CC * II;

  v8f acc[4] = {};
  const __bf16* Arow = act + (size_t)(m0 + lx) * II + half * 16;
  for (int k0 = 0; k0 < II; k0 += 32) {
    v16bf a = make_afrag(Arow + k0);
    if (k0 + 32 < II)
      __builtin_prefetch(W + (size_t)(n0 + lx) * II + k0 + 32, 0, 1);
#pragma unroll
    for (int t = 0; t < 4; ++t) {
      const __bf16* Br = W + (size_t)(n0 + t * 16 + lx) * II + k0 + half * 16;
      acc[t] = wmma_bf16(a, *(const v16bf*)Br, acc[t]);
    }
  }
#pragma unroll
  for (int t = 0; t < 4; ++t) {
    int col = n0 + t * 16 + lx;
#pragma unroll
    for (int r = 0; r < 8; ++r) {
      size_t idx = (size_t)(m0 + half * 8 + r) * CC + col;
      out[idx] = h[idx] + acc[t][r];
    }
  }
}

// ---------------------------------------------------------------------------
// Host-side launcher
// ---------------------------------------------------------------------------
extern "C" void kernel_launch(void* const* d_in, const int* in_sizes, int n_in,
                              void* d_out, int out_size, void* d_ws,
                              size_t ws_size, hipStream_t stream) {
  const float* x      = (const float*)d_in[0];
  const int*   ipos   = (const int*)d_in[1];
  const int*   tpos   = (const int*)d_in[2];
  const float* qkv_w  = (const float*)d_in[3];
  const float* qkv_b  = (const float*)d_in[4];
  const float* proj_w = (const float*)d_in[5];
  const float* proj_b = (const float*)d_in[6];
  const float* qn_w   = (const float*)d_in[7];
  const float* kn_w   = (const float*)d_in[8];
  const float* ln1_w  = (const float*)d_in[9];
  const float* ln2_w  = (const float*)d_in[10];
  const float* gate_w = (const float*)d_in[11];
  const float* up_w   = (const float*)d_in[12];
  const float* down_w = (const float*)d_in[13];
  float* out = (float*)d_out;

  char* ws = (char*)d_ws;
  size_t off = 0;
  auto carve = [&](size_t bytes) -> char* {
    char* p = ws + off;
    off = (off + bytes + 255) & ~(size_t)255;
    return p;
  };
  __bf16* qkvwT  = (__bf16*)carve((size_t)3 * C3 * CC * 2);
  __bf16* projwT = (__bf16*)carve((size_t)3 * CC * CC * 2);
  __bf16* gatewT = (__bf16*)carve((size_t)3 * II * CC * 2);
  __bf16* upwT   = (__bf16*)carve((size_t)3 * II * CC * 2);
  __bf16* downwT = (__bf16*)carve((size_t)3 * CC * II * 2);
  __bf16* xn     = (__bf16*)carve((size_t)MM * CC * 2);
  float*  qkvbuf = (float*)carve((size_t)MM * C3 * 4);
  __bf16* qb     = (__bf16*)carve((size_t)BB * HH * NN * DH * 2);
  __bf16* kb     = (__bf16*)carve((size_t)BB * HH * NN * DH * 2);
  __bf16* vb     = (__bf16*)carve((size_t)BB * HH * NN * DH * 2);
  __bf16* obuf   = (__bf16*)carve((size_t)MM * CC * 2);
  float*  hbuf   = (float*)carve((size_t)MM * CC * 4);
  __bf16* hn     = (__bf16*)carve((size_t)MM * CC * 2);
  __bf16* actb   = (__bf16*)carve((size_t)MM * II * 2);
  (void)ws_size; (void)n_in; (void)in_sizes; (void)out_size;

  auto tgrid = [](size_t elems) { return dim3((unsigned)((elems + 255) / 256)); };

  // 1) weights -> bf16, transposed so B-fragments load contiguously
  k_transpose_bf16<<<tgrid((size_t)3 * CC * C3), 256, 0, stream>>>(qkv_w,  qkvwT,  CC, C3);
  k_transpose_bf16<<<tgrid((size_t)3 * CC * CC), 256, 0, stream>>>(proj_w, projwT, CC, CC);
  k_transpose_bf16<<<tgrid((size_t)3 * CC * II), 256, 0, stream>>>(gate_w, gatewT, CC, II);
  k_transpose_bf16<<<tgrid((size_t)3 * CC * II), 256, 0, stream>>>(up_w,   upwT,   CC, II);
  k_transpose_bf16<<<tgrid((size_t)3 * II * CC), 256, 0, stream>>>(down_w, downwT, II, CC);

  // 2) RMSNorm(ln1)
  k_rmsnorm<<<dim3(MM), 256, 0, stream>>>(x, ln1_w, xn);

  // 3) QKV GEMM
  k_gemm_qkv<<<dim3(C3 / 64, MM / 16), 32, 0, stream>>>(xn, qkvwT, qkv_b, qkvbuf);

  // 4) q/k head RMSNorm + RoPE, split into per-head bf16 buffers
  k_qk_rope<<<dim3(BB * HH * NN), 32, 0, stream>>>(qkvbuf, ipos, tpos, qn_w,
                                                   kn_w, qb, kb, vb);

  // 5) flash attention with skip-causal mask (TDM-staged V tiles)
  k_attention<<<dim3(BB * HH * (NN / 16)), 32, 0, stream>>>(qb, kb, vb, obuf);

  // 6) proj GEMM + residual
  k_gemm_proj<<<dim3(CC / 64, MM / 16), 32, 0, stream>>>(obuf, projwT, proj_b,
                                                         x, hbuf);

  // 7) RMSNorm(ln2)
  k_rmsnorm<<<dim3(MM), 256, 0, stream>>>(hbuf, ln2_w, hn);

  // 8) gate/up GEMM + SiLU
  k_gemm_gateup<<<dim3(II / 64, MM / 16), 32, 0, stream>>>(hn, gatewT, upwT, actb);

  // 9) down GEMM + residual -> final output
  k_gemm_down<<<dim3(CC / 64, MM / 16), 32, 0, stream>>>(actb, downwT, hbuf, out);
}
